// GraphConvolutionalNetwork_57415122812990
// MI455X (gfx1250) — matrix operand order repaired and checked by reading, not compiled, not run on hardware
//
#include <hip/hip_runtime.h>
#include <hip/hip_bf16.h>

// ---------------- problem constants ----------------
#define GN   100000   // nodes
#define GE   1600000  // edges
#define GIN  128      // in feats
#define GHID 256      // hidden
#define GOUT 160      // output feats
#define GEPS 1e-5f

#define NP   100096   // GN rounded up to multiple of 128 (padded rows)

typedef __attribute__((ext_vector_type(16))) __bf16 v16bf;
typedef __attribute__((ext_vector_type(8)))  __bf16 v8bf;
typedef __attribute__((ext_vector_type(8)))  float  v8f;

// ---------------- workspace layout (bytes) ----------------
// all regions 1KB aligned
static constexpr size_t OFF_SSUM = 0;                                   // NP*GIN f32
static constexpr size_t OFF_DEG  = OFF_SSUM + (size_t)NP * GIN * 4;     // NP f32
static constexpr size_t OFF_SUM  = OFF_DEG  + (size_t)NP * 4;           // 256 f32
static constexpr size_t OFF_SQ   = OFF_SUM  + 1024;                     // 256 f32
static constexpr size_t OFF_WST  = OFF_SQ   + 1024;                     // GHID*GIN bf16 (W_self^T)
static constexpr size_t OFF_WNT  = OFF_WST  + (size_t)GHID * GIN * 2;   // GHID*GIN bf16 (W_neigh^T)
static constexpr size_t OFF_W2T  = OFF_WNT  + (size_t)GHID * GIN * 2;   // GOUT*GHID bf16 (folded fc^T)
static constexpr size_t OFF_B2   = OFF_W2T  + (size_t)GOUT * GHID * 2;  // GOUT f32
static constexpr size_t OFF_RST  = OFF_B2   + 1024;                     // NP*GHID bf16

// async copy 16B from global to LDS (per-lane addresses), tracked by ASYNCcnt
__device__ __forceinline__ void async_b128(const void* gptr, void* ldsptr) {
    uint32_t l = (uint32_t)(uintptr_t)ldsptr;          // low 32 bits = LDS offset
    uint64_t g = (uint64_t)(uintptr_t)gptr;
    asm volatile("global_load_async_to_lds_b128 %0, %1, off"
                 :: "v"(l), "v"(g) : "memory");
}
__device__ __forceinline__ void wait_async0() {
    asm volatile("s_wait_asynccnt 0x0" ::: "memory");
}

// ---------------- kernel 1: zero scratch ----------------
__global__ __launch_bounds__(256) void k_zero(float* __restrict__ p, size_t n4) {
    size_t i = (size_t)blockIdx.x * blockDim.x + threadIdx.x;
    size_t stride = (size_t)gridDim.x * blockDim.x;
    float4 z = make_float4(0.f, 0.f, 0.f, 0.f);
    for (; i < n4; i += stride) ((float4*)p)[i] = z;
}

// ---------------- kernel 2: weights -> bf16 transposed ----------------
__global__ __launch_bounds__(256) void k_prepw(const float* __restrict__ Ws,
                                               const float* __restrict__ Wn,
                                               __bf16* __restrict__ WsT,
                                               __bf16* __restrict__ WnT) {
    int idx = blockIdx.x * 256 + threadIdx.x;        // idx = c*GIN + k
    if (idx >= GHID * GIN) return;
    int c = idx >> 7;
    int k = idx & (GIN - 1);
    WsT[idx] = (__bf16)Ws[k * GHID + c];
    WnT[idx] = (__bf16)Wn[k * GHID + c];
}

// ---------------- kernel 3: edge scatter (one wave32 per edge) ----------------
__global__ __launch_bounds__(256) void k_scatter(const float* __restrict__ feat,
                                                 const int* __restrict__ src,
                                                 const int* __restrict__ dst,
                                                 const float* __restrict__ ew,
                                                 float* __restrict__ ssum,
                                                 float* __restrict__ deg) {
    int e = blockIdx.x * 8 + (threadIdx.x >> 5);
    if (e >= GE) return;
    int lane = threadIdx.x & 31;
    int s = src[e];
    int d = dst[e];
    float w = ew[e];
    float4 f = *(const float4*)(feat + (size_t)s * GIN + lane * 4);
    float* p = ssum + (size_t)d * GIN + lane * 4;
    unsafeAtomicAdd(p + 0, f.x * w);
    unsafeAtomicAdd(p + 1, f.y * w);
    unsafeAtomicAdd(p + 2, f.z * w);
    unsafeAtomicAdd(p + 3, f.w * w);
    if (lane == 0) unsafeAtomicAdd(deg + d, 1.0f);
}

// ---------------- kernel 4: fused GEMM1 + ReLU + BN stats ----------------
// rst = relu(X @ Ws + (ssum/deg) @ Wn + bias) ; accumulate per-channel sum/sumsq
#define LDA1 136   // 128 + 8-half pad (272B row stride)
__global__ __launch_bounds__(256) void k_gemm1(const float* __restrict__ feat,
                                               const float* __restrict__ ssum,
                                               const float* __restrict__ deg,
                                               const __bf16* __restrict__ WsT,
                                               const __bf16* __restrict__ WnT,
                                               const float* __restrict__ bias,
                                               __bf16* __restrict__ rst,
                                               float* __restrict__ stat_sum,
                                               float* __restrict__ stat_sq) {
    __shared__ __align__(16) __bf16 Xs[128 * LDA1];
    __shared__ __align__(16) __bf16 Gs[128 * LDA1];
    const int R = blockIdx.x * 128;
    const int t = threadIdx.x;

    // ---- stage X and G tiles (f32 -> bf16) into LDS ----
    for (int idx = t; idx < 128 * 32; idx += 256) {
        int row = idx >> 5;
        int c4  = (idx & 31) * 4;
        int grow = R + row;
        float4 f = make_float4(0.f, 0.f, 0.f, 0.f);
        if (grow < GN) f = *(const float4*)(feat + (size_t)grow * GIN + c4);
        __bf16 xb[4] = { (__bf16)f.x, (__bf16)f.y, (__bf16)f.z, (__bf16)f.w };
        *(uint2*)&Xs[row * LDA1 + c4] = *(uint2*)xb;
        // padded ssum/deg rows are zeroed, safe to read unconditionally
        float dg  = deg[grow];
        float inv = 1.0f / fmaxf(dg, 1.0f);
        float4 g  = *(const float4*)(ssum + (size_t)grow * GIN + c4);
        __bf16 gb[4] = { (__bf16)(g.x * inv), (__bf16)(g.y * inv),
                         (__bf16)(g.z * inv), (__bf16)(g.w * inv) };
        *(uint2*)&Gs[row * LDA1 + c4] = *(uint2*)gb;
    }
    __syncthreads();

    const int w      = t >> 5;          // wave id: rows [w*16, w*16+16)
    const int lane   = t & 31;
    const int mrow   = w * 16 + (lane & 15);
    const int kbA    = (lane >> 4) * 16;   // A frag: K 0..15 / 16..31
    const int nlane  = lane & 15;
    const int khalfB = (lane >> 4) * 16;   // B frag: K 0..15 / 16..31

    for (int nb = 0; nb < 4; ++nb) {       // 64 output cols per nb block
        v8f acc[4];
        #pragma unroll
        for (int tt = 0; tt < 4; ++tt) acc[tt] = (v8f){0.f,0.f,0.f,0.f,0.f,0.f,0.f,0.f};

        #pragma unroll
        for (int kk = 0; kk < GIN; kk += 32) {
            v8bf xlo = *(const v8bf*)&Xs[mrow * LDA1 + kk + kbA];
            v8bf xhi = *(const v8bf*)&Xs[mrow * LDA1 + kk + kbA + 8];
            v8bf glo = *(const v8bf*)&Gs[mrow * LDA1 + kk + kbA];
            v8bf ghi = *(const v8bf*)&Gs[mrow * LDA1 + kk + kbA + 8];
            v16bf ax, ag;
            #pragma unroll
            for (int i = 0; i < 8; ++i) {
                ax[i] = xlo[i]; ax[i + 8] = xhi[i];
                ag[i] = glo[i]; ag[i + 8] = ghi[i];
            }
            #pragma unroll
            for (int tt = 0; tt < 4; ++tt) {
                int n = nb * 64 + tt * 16 + nlane;
                v16bf bs = *(const v16bf*)(WsT + (size_t)n * GIN + kk + khalfB);
                v16bf bn = *(const v16bf*)(WnT + (size_t)n * GIN + kk + khalfB);
                acc[tt] = __builtin_amdgcn_wmma_f32_16x16x32_bf16(
                              false, ax, false, bs, (short)0, acc[tt], false, false);
                acc[tt] = __builtin_amdgcn_wmma_f32_16x16x32_bf16(
                              false, ag, false, bn, (short)0, acc[tt], false, false);
            }
        }

        // ---- epilogue: bias + relu, store bf16, accumulate BN stats ----
        #pragma unroll
        for (int tt = 0; tt < 4; ++tt) {
            int ch = nb * 64 + tt * 16 + nlane;
            float bv = bias[ch];
            float lsum = 0.f, lsq = 0.f;
            #pragma unroll
            for (int v = 0; v < 8; ++v) {
                int rm   = v + 8 * (lane >> 4);
                int grow = R + w * 16 + rm;
                float val = fmaxf(acc[tt][v] + bv, 0.0f);
                if (grow < GN) {
                    lsum += val;
                    lsq  += val * val;
                    rst[(size_t)grow * GHID + ch] = (__bf16)val;
                }
            }
            unsafeAtomicAdd(stat_sum + ch, lsum);
            unsafeAtomicAdd(stat_sq + ch, lsq);
        }
    }
}

// ---------------- kernel 5: fold BN into fc ----------------
__global__ __launch_bounds__(256) void k_bnfold(const float* __restrict__ gamma,
                                                const float* __restrict__ beta,
                                                const float* __restrict__ fcW,
                                                const float* __restrict__ fcb,
                                                const float* __restrict__ stat_sum,
                                                const float* __restrict__ stat_sq,
                                                __bf16* __restrict__ W2T,
                                                float* __restrict__ b2) {
    __shared__ float aS[GHID], bS[GHID];
    int t = threadIdx.x;
    if (t < GHID) {
        float mu  = stat_sum[t] * (1.0f / GN);
        float var = stat_sq[t] * (1.0f / GN) - mu * mu;
        float a   = gamma[t] * rsqrtf(var + GEPS);
        aS[t] = a;
        bS[t] = beta[t] - mu * a;
    }
    __syncthreads();
    for (int idx = t; idx < GOUT * GHID; idx += 256) {   // idx = o*GHID + c
        int o = idx >> 8;
        int c = idx & (GHID - 1);
        W2T[idx] = (__bf16)(aS[c] * fcW[c * GOUT + o]);
    }
    for (int o = t; o < GOUT; o += 256) {
        float s = fcb[o];
        for (int c = 0; c < GHID; ++c) s += bS[c] * fcW[c * GOUT + o];
        b2[o] = s;
    }
}

// ---------------- kernel 6: GEMM2 out = rst @ W2 + b2 ----------------
// A tile and full folded-weight matrix staged into LDS via async-to-LDS path;
// inner loop is pure ds_load + wmma.
#define LDA2 264   // 256 + 8-half pad (528B row stride)
#define LDB2 264   // 256 + 8-half pad for weights (bank spread)
__global__ __launch_bounds__(256) void k_gemm2(const __bf16* __restrict__ rst,
                                               const __bf16* __restrict__ W2T,
                                               const float* __restrict__ b2,
                                               float* __restrict__ out) {
    __shared__ __align__(16) __bf16 As[128 * LDA2];     //  66.0 KB
    __shared__ __align__(16) __bf16 W2s[GOUT * LDB2];   //  82.5 KB
    const int R = blockIdx.x * 128;
    const int t = threadIdx.x;

    // ---- async stage 128x256 bf16 A tile (rst padded to NP rows, safe) ----
    for (int idx = t; idx < 128 * 32; idx += 256) {
        int row = idx >> 5;
        int c8  = (idx & 31) * 8;
        async_b128(rst + (size_t)(R + row) * GHID + c8, &As[row * LDA2 + c8]);
    }
    // ---- async stage 160x256 bf16 folded weights ----
    for (int idx = t; idx < GOUT * 32; idx += 256) {
        int o  = idx >> 5;
        int c8 = (idx & 31) * 8;
        async_b128(W2T + (size_t)o * GHID + c8, &W2s[o * LDB2 + c8]);
    }
    wait_async0();
    __syncthreads();

    const int w      = t >> 5;
    const int lane   = t & 31;
    const int mrow   = w * 16 + (lane & 15);
    const int kbA    = (lane >> 4) * 8;
    const int nlane  = lane & 15;
    const int khalfB = (lane >> 4) * 8;

    v8f acc[10];
    #pragma unroll
    for (int tt = 0; tt < 10; ++tt) acc[tt] = (v8f){0.f,0.f,0.f,0.f,0.f,0.f,0.f,0.f};

    #pragma unroll
    for (int kk = 0; kk < GHID; kk += 32) {
        v8bf alo = *(const v8bf*)&As[mrow * LDA2 + kk + kbA];
        v8bf ahi = *(const v8bf*)&As[mrow * LDA2 + kk + kbA + 16];
        v16bf a;
        #pragma unroll
        for (int i = 0; i < 8; ++i) { a[i] = alo[i]; a[i + 8] = ahi[i]; }
        #pragma unroll
        for (int tt = 0; tt < 10; ++tt) {
            int n = tt * 16 + nlane;
            v8bf blo = *(const v8bf*)&W2s[n * LDB2 + kk + khalfB];
            v8bf bhi = *(const v8bf*)&W2s[n * LDB2 + kk + khalfB + 16];
            v16bf b;
            #pragma unroll
            for (int i = 0; i < 8; ++i) { b[i] = blo[i]; b[i + 8] = bhi[i]; }
            acc[tt] = __builtin_amdgcn_wmma_f32_16x16x32_bf16(
                          false, a, false, b, (short)0, acc[tt], false, false);
        }
    }

    #pragma unroll
    for (int tt = 0; tt < 10; ++tt) {
        int col = tt * 16 + nlane;
        float bb = b2[col];
        #pragma unroll
        for (int v = 0; v < 8; ++v) {
            int grow = R + w * 16 + v + 8 * (lane >> 4);
            if (grow < GN) out[(size_t)grow * GOUT + col] = acc[tt][v] + bb;
        }
    }
}

// ---------------- launcher ----------------
extern "C" void kernel_launch(void* const* d_in, const int* in_sizes, int n_in,
                              void* d_out, int out_size, void* d_ws, size_t ws_size,
                              hipStream_t stream) {
    const float* feat  = (const float*)d_in[0];
    const int*   src   = (const int*)d_in[1];
    const int*   dst   = (const int*)d_in[2];
    const float* ew    = (const float*)d_in[3];
    const float* Ws    = (const float*)d_in[4];
    const float* Wn    = (const float*)d_in[5];
    const float* bias  = (const float*)d_in[6];
    const float* gamma = (const float*)d_in[7];
    const float* beta  = (const float*)d_in[8];
    const float* fcW   = (const float*)d_in[9];
    const float* fcb   = (const float*)d_in[10];
    float* out = (float*)d_out;

    char* ws = (char*)d_ws;
    float*  ssum     = (float*)(ws + OFF_SSUM);
    float*  deg      = (float*)(ws + OFF_DEG);
    float*  stat_sum = (float*)(ws + OFF_SUM);
    float*  stat_sq  = (float*)(ws + OFF_SQ);
    __bf16* WsT      = (__bf16*)(ws + OFF_WST);
    __bf16* WnT      = (__bf16*)(ws + OFF_WNT);
    __bf16* W2T      = (__bf16*)(ws + OFF_W2T);
    float*  b2       = (float*)(ws + OFF_B2);
    __bf16* rst      = (__bf16*)(ws + OFF_RST);

    // zero ssum + deg + stats (contiguous region)
    size_t zero_f  = (size_t)NP * GIN + NP + 512;
    size_t zero_f4 = zero_f / 4;
    k_zero<<<2048, 256, 0, stream>>>(ssum, zero_f4);

    k_prepw<<<(GHID * GIN + 255) / 256, 256, 0, stream>>>(Ws, Wn, WsT, WnT);

    k_scatter<<<(GE + 7) / 8, 256, 0, stream>>>(feat, src, dst, ew, ssum, deg);

    int nblk = (GN + 127) / 128;
    k_gemm1<<<nblk, 256, 0, stream>>>(feat, ssum, deg, WsT, WnT, bias,
                                      rst, stat_sum, stat_sq);

    k_bnfold<<<1, 256, 0, stream>>>(gamma, beta, fcW, fcb, stat_sum, stat_sq, W2T, b2);

    k_gemm2<<<nblk, 256, 0, stream>>>(rst, W2T, b2, out);
}
